// Attention_65712999629234
// MI455X (gfx1250) — matrix operand order repaired and checked by reading, not compiled, not run on hardware
//
#include <hip/hip_runtime.h>
#include <hip/hip_bf16.h>

typedef __bf16 bf16;
typedef __attribute__((ext_vector_type(16))) __bf16 v16bf;
typedef __attribute__((ext_vector_type(8)))  float  v8f;
typedef __attribute__((ext_vector_type(4)))  unsigned int v4u;
typedef __attribute__((ext_vector_type(8)))  int  v8i;
typedef __attribute__((ext_vector_type(4)))  int  v4i;

#define SEQ     4096
#define HIDDEN  640
#define HD      80
#define DPAD    96
#define NH      8
#define NKV     2
#define NREP    4
#define WINDOW  512
#define SINK    4
#define KV_DIM  160

#if defined(__HIP_DEVICE_COMPILE__) && __has_builtin(__builtin_amdgcn_tensor_load_to_lds)
#define HAVE_TDM 1
#else
#define HAVE_TDM 0
#endif

// ---------------------------------------------------------------------------
// WMMA helpers (CDNA5 16x16x32 bf16, wave32)
// ---------------------------------------------------------------------------
__device__ inline v8f wmma_bf16(v16bf a, v16bf b, v8f c) {
  return __builtin_amdgcn_wmma_f32_16x16x32_bf16(
      /*neg_a=*/false, a, /*neg_b=*/false, b,
      /*c_mod=*/(short)0, c, /*reuse_a=*/false, /*reuse_b=*/false);
}

// A-matrix (16x32, row-major source): lane l holds row m=l&15,
// k = (l>>4)*8 + (e>>3)*16 + (e&7)  -> two contiguous 8-elem (16B) runs.
__device__ inline v16bf load_a16(const bf16* base, int stride, int lane) {
  const int m  = lane & 15;
  const int k0 = (lane >> 4) << 3;
  const bf16* p = base + (size_t)m * stride + k0;
  v16bf a;
#pragma unroll
  for (int e = 0; e < 8; ++e) a[e] = p[e];
#pragma unroll
  for (int e = 0; e < 8; ++e) a[8 + e] = p[16 + e];
  return a;
}

// B-matrix (32x16) from a column-contiguous source (row n holds its K run):
// lane l holds column n=l&15, k = (l>>4)*8 + (e>>3)*16 + (e&7)  -> two contiguous 8-elem (16B) runs.
__device__ inline v16bf load_bT16(const bf16* base, int stride, int lane) {
  const int n  = lane & 15;
  const int k0 = (lane >> 4) << 3;
  const bf16* p = base + (size_t)n * stride + k0;
  v16bf b;
#pragma unroll
  for (int e = 0; e < 8; ++e) b[e] = p[e];
#pragma unroll
  for (int e = 0; e < 8; ++e) b[8 + e] = p[16 + e];
  return b;
}

__device__ inline void store_c16(float* __restrict__ C, int N, int tm, int tn,
                                 int lane, v8f c) {
  const int n  = tn + (lane & 15);
  const int m0 = tm + ((lane >> 4) << 3);
#pragma unroll
  for (int r = 0; r < 8; ++r) C[(size_t)(m0 + r) * N + n] = c[r];
}

#if HAVE_TDM
// Issue a 2D TDM tile load: global (row-major, element=2B) -> LDS.
// dims/strides in 2-byte elements. D# packed per CDNA5 ISA 8.3/8.4.
__device__ inline void tdm_load_2d(unsigned lds_off, unsigned long long gaddr,
                                   unsigned td0, unsigned td1,
                                   unsigned tl0, unsigned tl1, unsigned st0) {
  v4u g0;
  g0[0] = 1u;                                   // count=1, user mode
  g0[1] = lds_off;                              // lds_addr (bytes)
  g0[2] = (unsigned)gaddr;                      // global_addr[31:0]
  g0[3] = (unsigned)((gaddr >> 32) & 0x01FFFFFFu) | (2u << 30);  // [56:32]|type=2
  v8i g1;
  g1[0] = 0x00010000;                            // mask=0, data_size=1 (2B)
  g1[1] = (int)(td0 << 16);                      // tensor_dim0[15:0] @ bits31:16
  g1[2] = (int)((td0 >> 16) | (td1 << 16));      // tdim0[31:16] | tdim1[15:0]
  g1[3] = (int)((td1 >> 16) | (tl0 << 16));      // tdim1[31:16] | tile_dim0
  g1[4] = (int)(tl1 & 0xFFFFu);                  // tile_dim1 | tile_dim2=0
  g1[5] = (int)st0;                              // tensor_dim0_stride[31:0]
  g1[6] = 0;
  g1[7] = 0;
  v4i gz = {0, 0, 0, 0};
#if defined(__clang_major__) && (__clang_major__ >= 23)
  v8i gz8 = {0, 0, 0, 0, 0, 0, 0, 0};
  __builtin_amdgcn_tensor_load_to_lds(g0, g1, gz, gz, gz8, 0);
#else
  __builtin_amdgcn_tensor_load_to_lds(g0, g1, gz, gz, 0);
#endif
}
#endif

// ---------------------------------------------------------------------------
// WMMA GEMM, 64x32 output tile per wave: 8 WMMAs per (4 A + 2 B) loads per
// K-step. C[MxN] f32 = A[MxK] bf16 (row-major) @ B, B given as BT[NxK] bf16.
// ---------------------------------------------------------------------------
__global__ void gemm_wmma(const bf16* __restrict__ A, const bf16* __restrict__ BT,
                          float* __restrict__ C, int M, int N, int K) {
  const int lane = threadIdx.x & 31;
  const int wave = threadIdx.x >> 5;
  const int tilesN = N >> 5;
  const int tile = blockIdx.x * (blockDim.x >> 5) + wave;
  if (tile >= (M >> 6) * tilesN) return;  // wave-uniform
  const int tm = (tile / tilesN) << 6;
  const int tn = (tile % tilesN) << 5;
  v8f acc[4][2];
#pragma unroll
  for (int i = 0; i < 4; ++i)
#pragma unroll
    for (int j = 0; j < 2; ++j)
#pragma unroll
      for (int r = 0; r < 8; ++r) acc[i][j][r] = 0.f;

  const bf16* brow0 = BT + (size_t)tn * K;
  const bf16* brow1 = BT + (size_t)(tn + 16) * K;
  for (int k = 0; k < K; k += 32) {
    v16bf a[4];
#pragma unroll
    for (int i = 0; i < 4; ++i)
      a[i] = load_a16(A + (size_t)(tm + 16 * i) * K + k, K, lane);
    v16bf b0 = load_bT16(brow0 + k, K, lane);
    v16bf b1 = load_bT16(brow1 + k, K, lane);
#pragma unroll
    for (int i = 0; i < 4; ++i) {
      acc[i][0] = wmma_bf16(a[i], b0, acc[i][0]);
      acc[i][1] = wmma_bf16(a[i], b1, acc[i][1]);
    }
  }
#pragma unroll
  for (int i = 0; i < 4; ++i) {
    store_c16(C, N, tm + 16 * i, tn,      lane, acc[i][0]);
    store_c16(C, N, tm + 16 * i, tn + 16, lane, acc[i][1]);
  }
}

// ---------------------------------------------------------------------------
// Conversion / transpose kernels
// ---------------------------------------------------------------------------
__global__ void f32_to_bf16(const float* __restrict__ in, bf16* __restrict__ out, int n) {
  int i = blockIdx.x * blockDim.x + threadIdx.x;
  if (i < n) out[i] = (bf16)in[i];
}

// W[K][N] f32 -> WT[N][K] bf16
__global__ void transpose_to_bf16(const float* __restrict__ W, bf16* __restrict__ WT,
                                  int K, int N) {
  int i = blockIdx.x * blockDim.x + threadIdx.x;
  if (i < K * N) {
    int k = i % K, n = i / K;
    WT[i] = (bf16)W[(size_t)k * N + n];
  }
}

// ---------------------------------------------------------------------------
// RMSNorm + RoPE (+ V convert). One wave per (s, row).
// blockIdx.y: 0..7 Q heads, 8..9 K kv-heads, 10..11 V kv-heads.
// Q/K padded to DPAD=96; V written TRANSPOSED to global: Vt[kvh][dim][seq]
// so the attention TDM can gather a dim-major tile with stride0 = SEQ.
// ---------------------------------------------------------------------------
__global__ void normrope(const float* __restrict__ Qraw, const float* __restrict__ Kraw,
                         const float* __restrict__ Vraw,
                         const float* __restrict__ cosb, const float* __restrict__ sinb,
                         const float* __restrict__ qw, const float* __restrict__ kw,
                         bf16* __restrict__ Qb, bf16* __restrict__ Kb,
                         bf16* __restrict__ Vt) {
  const int s = blockIdx.x;
  const int idx = blockIdx.y;
  const int lane = threadIdx.x;

  if (idx >= 10) {  // V: convert + transpose to [kvh][dim][seq]
    const int kvh = idx - 10;
    const float* src = Vraw + (size_t)s * KV_DIM + kvh * HD;
    bf16* dst = Vt + (size_t)kvh * HD * SEQ + s;
    for (int d = lane; d < HD; d += 32) dst[(size_t)d * SEQ] = (bf16)src[d];
    return;
  }

  const float* src;
  const float* w;
  bf16* dst;
  if (idx < NH) {
    src = Qraw + (size_t)s * HIDDEN + idx * HD;
    w = qw;
    dst = Qb + ((size_t)idx * SEQ + s) * DPAD;
  } else {
    const int kvh = idx - NH;
    src = Kraw + (size_t)s * KV_DIM + kvh * HD;
    w = kw;
    dst = Kb + ((size_t)kvh * SEQ + s) * DPAD;
  }

  float ss = 0.f;
  for (int d = lane; d < HD; d += 32) { float v = src[d]; ss += v * v; }
#pragma unroll
  for (int off = 16; off > 0; off >>= 1) ss += __shfl_xor(ss, off, 32);
  const float rinv = rsqrtf(ss / (float)HD + 1e-5f);

  for (int p = lane; p < HD / 2; p += 32) {
    float ve = src[2 * p]     * rinv * w[2 * p];
    float vo = src[2 * p + 1] * rinv * w[2 * p + 1];
    float c  = cosb[(size_t)s * (HD / 2) + p];
    float sn = sinb[(size_t)s * (HD / 2) + p];
    dst[2 * p]     = (bf16)(ve * c - vo * sn);
    dst[2 * p + 1] = (bf16)(vo * c + ve * sn);
  }
  for (int d = HD + lane; d < DPAD; d += 32) dst[d] = (bf16)0.0f;
}

// ---------------------------------------------------------------------------
// Flash attention with sliding window + sink, all matmuls through WMMA.
// Per 32-key chunk, the TDM stages BOTH the K tile (32x96, row-major) and the
// V^T tile (80 dims x 32 keys, stride SEQ) into LDS; all subsequent WMMA
// operand reads from LDS are two 16B runs per lane.
// ---------------------------------------------------------------------------
__global__ void attn_swa(const bf16* __restrict__ Qb, const bf16* __restrict__ Kb,
                         const bf16* __restrict__ Vt, bf16* __restrict__ AOb) {
  __shared__ bf16 Klds[32 * DPAD];      // K chunk: [key][dim] row-major
  __shared__ bf16 VldsT[HD * 32];       // V chunk: [dim][key] dim-major
  __shared__ bf16 Plds[4][16 * 32];     // per-wave P re-layout staging

  const int h    = blockIdx.x;
  const int kvh  = h / NREP;
  const int qb   = blockIdx.y * 64;
  const int tid  = threadIdx.x;
  const int wave = tid >> 5;
  const int lane = tid & 31;
  const int q0   = qb + wave * 16;
  const int hl   = lane >> 4;
  const int ln   = lane & 15;
  const float scale = 0.1118033988749895f;  // 1/sqrt(80)

  // Preload Q A-tiles (16 queries x 96 dims = 3 chunks of 32)
  const bf16* qbase = Qb + ((size_t)h * SEQ + q0) * DPAD;
  v16bf aq0 = load_a16(qbase,      DPAD, lane);
  v16bf aq1 = load_a16(qbase + 32, DPAD, lane);
  v16bf aq2 = load_a16(qbase + 64, DPAD, lane);

  float mstate[8], lstate[8], corr[8];
  v8f acc[5];
#pragma unroll
  for (int r = 0; r < 8; ++r) { mstate[r] = -1e30f; lstate[r] = 0.f; }
#pragma unroll
  for (int d = 0; d < 5; ++d)
#pragma unroll
    for (int r = 0; r < 8; ++r) acc[d][r] = 0.f;

  const int wstart = (qb - (WINDOW - 1)) > 0 ? (qb - (WINDOW - 1)) : 0;
  const int cend = qb + 63;

#if HAVE_TDM
  const unsigned klds_off = (unsigned)(unsigned long long)(&Klds[0]);
  const unsigned vlds_off = (unsigned)(unsigned long long)(&VldsT[0]);
#endif

  for (int kc = 0; kc <= cend; kc += 32) {
    // keep chunk 0 (sinks) and chunks overlapping the block's union window
    if (kc > 0 && kc + 31 < wstart) continue;  // block-uniform

    // ---- Stage K (32x96) and V^T (80x32) chunks into LDS ----
#if HAVE_TDM
    if (wave == 0) {
      const unsigned long long kga =
          (unsigned long long)(const void*)(Kb + ((size_t)kvh * SEQ + kc) * DPAD);
      const unsigned long long vga =
          (unsigned long long)(const void*)(Vt + (size_t)kvh * HD * SEQ + kc);
      // K: tensor 96 x SEQ rows, tile 96x32, stride0=96 (contiguous rows)
      tdm_load_2d(klds_off, kga, DPAD, SEQ, DPAD, 32, DPAD);
      // V^T: tensor SEQ x HD rows, tile 32x80, stride0=SEQ (strided gather)
      tdm_load_2d(vlds_off, vga, SEQ, HD, 32, HD, SEQ);
      __builtin_amdgcn_s_wait_tensorcnt((short)0);
    }
#else
    {
      const bf16* ksrc = Kb + ((size_t)kvh * SEQ + kc) * DPAD;
      for (int i = tid; i < 32 * DPAD; i += 128) Klds[i] = ksrc[i];
      const bf16* vsrc = Vt + (size_t)kvh * HD * SEQ + kc;
      for (int i = tid; i < HD * 32; i += 128) {
        int d = i >> 5, kk = i & 31;
        VldsT[i] = vsrc[(size_t)d * SEQ + kk];
      }
    }
#endif
    __syncthreads();

    // S = Q K^T for 2 key sub-tiles of 16, K-dim 96 in 3 WMMA steps.
    // B operand from Klds: lane=key row, contiguous dims.
    v8f s0 = {0.f, 0.f, 0.f, 0.f, 0.f, 0.f, 0.f, 0.f};
    v8f s1 = {0.f, 0.f, 0.f, 0.f, 0.f, 0.f, 0.f, 0.f};
    {
      const bf16* kb0 = &Klds[0];
      const bf16* kb1 = &Klds[16 * DPAD];
      v16bf b;
      b = load_bT16(kb0,      DPAD, lane); s0 = wmma_bf16(aq0, b, s0);
      b = load_bT16(kb0 + 32, DPAD, lane); s0 = wmma_bf16(aq1, b, s0);
      b = load_bT16(kb0 + 64, DPAD, lane); s0 = wmma_bf16(aq2, b, s0);
      b = load_bT16(kb1,      DPAD, lane); s1 = wmma_bf16(aq0, b, s1);
      b = load_bT16(kb1 + 32, DPAD, lane); s1 = wmma_bf16(aq1, b, s1);
      b = load_bT16(kb1 + 64, DPAD, lane); s1 = wmma_bf16(aq2, b, s1);
    }

    // Mask + online softmax. C layout: lane=key(ln), VGPR r -> row m=r+hl*8.
#pragma unroll
    for (int r = 0; r < 8; ++r) {
      const int m  = r + (hl << 3);
      const int qi = q0 + m;
      const int k0i = kc + ln;
      const int k1i = kc + 16 + ln;
      float v0 = s0[r] * scale;
      float v1 = s1[r] * scale;
      const bool ok0 = (k0i <= qi) && ((k0i >= qi - (WINDOW - 1)) || (k0i < SINK));
      const bool ok1 = (k1i <= qi) && ((k1i >= qi - (WINDOW - 1)) || (k1i < SINK));
      if (!ok0) v0 = -1e30f;
      if (!ok1) v1 = -1e30f;
      float rm = fmaxf(v0, v1);
#pragma unroll
      for (int off = 1; off < 16; off <<= 1) rm = fmaxf(rm, __shfl_xor(rm, off, 32));
      const float newm = fmaxf(mstate[r], rm);
      const float cr = __expf(mstate[r] - newm);
      const float p0 = __expf(v0 - newm);
      const float p1 = __expf(v1 - newm);
      float psum = p0 + p1;
#pragma unroll
      for (int off = 1; off < 16; off <<= 1) psum += __shfl_xor(psum, off, 32);
      lstate[r] = lstate[r] * cr + psum;
      mstate[r] = newm;
      corr[r] = cr;
      Plds[wave][m * 32 + ln]      = (bf16)p0;
      Plds[wave][m * 32 + 16 + ln] = (bf16)p1;
    }
#pragma unroll
    for (int d = 0; d < 5; ++d)
#pragma unroll
      for (int r = 0; r < 8; ++r) acc[d][r] *= corr[r];

    // P was written cross-lane within this wave; drain LDS before A-load
    asm volatile("s_wait_dscnt 0" ::: "memory");
    v16bf ap = load_a16(&Plds[wave][0], 32, lane);

    // O += P @ V : 5 dim-tiles of 16, K-dim = 32 keys. B operand from VldsT:
    // lane = dim row, keys k0..k0+7 and k0+16..k0+23 -> two 16B LDS runs per lane.
#pragma unroll
    for (int d = 0; d < 5; ++d) {
      v16bf bv = load_bT16(&VldsT[(size_t)(d * 16) * 32], 32, lane);
      acc[d] = wmma_bf16(ap, bv, acc[d]);
    }
    __syncthreads();  // protect Klds/VldsT before next chunk overwrites them
  }

  // Normalize and store [s][h*80+d] in bf16 for the output projection
#pragma unroll
  for (int r = 0; r < 8; ++r) {
    const int m  = r + (hl << 3);
    const int qi = q0 + m;
    const float inv = 1.0f / lstate[r];
#pragma unroll
    for (int d = 0; d < 5; ++d)
      AOb[(size_t)qi * HIDDEN + h * HD + d * 16 + ln] = (bf16)(acc[d][r] * inv);
  }
}

// ---------------------------------------------------------------------------
// Host launcher
// ---------------------------------------------------------------------------
extern "C" void kernel_launch(void* const* d_in, const int* in_sizes, int n_in,
                              void* d_out, int out_size, void* d_ws, size_t ws_size,
                              hipStream_t stream) {
  (void)in_sizes; (void)n_in; (void)out_size; (void)ws_size;
  const float* x    = (const float*)d_in[0];
  const float* cosb = (const float*)d_in[1];
  const float* sinb = (const float*)d_in[2];
  const float* Wq   = (const float*)d_in[3];
  const float* Wk   = (const float*)d_in[4];
  const float* Wv   = (const float*)d_in[5];
  const float* Wo   = (const float*)d_in[6];
  const float* qw   = (const float*)d_in[7];
  const float* kw   = (const float*)d_in[8];
  float* out = (float*)d_out;

  char* ws = (char*)d_ws;
  size_t off = 0;
  auto alloc = [&](size_t bytes) -> void* {
    void* p = ws + off;
    off += (bytes + 255) & ~(size_t)255;
    return p;
  };
  bf16*  xb   = (bf16*)alloc((size_t)SEQ * HIDDEN * 2);
  bf16*  WqT  = (bf16*)alloc((size_t)HIDDEN * HIDDEN * 2);
  bf16*  WkT  = (bf16*)alloc((size_t)KV_DIM * HIDDEN * 2);
  bf16*  WvT  = (bf16*)alloc((size_t)KV_DIM * HIDDEN * 2);
  bf16*  WoT  = (bf16*)alloc((size_t)HIDDEN * HIDDEN * 2);
  float* Qraw = (float*)alloc((size_t)SEQ * HIDDEN * 4);
  float* Kraw = (float*)alloc((size_t)SEQ * KV_DIM * 4);
  float* Vraw = (float*)alloc((size_t)SEQ * KV_DIM * 4);
  bf16*  Qb   = (bf16*)alloc((size_t)NH * SEQ * DPAD * 2);
  bf16*  Kb   = (bf16*)alloc((size_t)NKV * SEQ * DPAD * 2);
  bf16*  Vt   = (bf16*)alloc((size_t)NKV * HD * SEQ * 2);
  bf16*  AOb  = (bf16*)alloc((size_t)SEQ * HIDDEN * 2);

  // 1) precision conversion / weight transposes
  f32_to_bf16<<<(SEQ * HIDDEN + 255) / 256, 256, 0, stream>>>(x, xb, SEQ * HIDDEN);
  transpose_to_bf16<<<(HIDDEN * HIDDEN + 255) / 256, 256, 0, stream>>>(Wq, WqT, HIDDEN, HIDDEN);
  transpose_to_bf16<<<(HIDDEN * KV_DIM + 255) / 256, 256, 0, stream>>>(Wk, WkT, HIDDEN, KV_DIM);
  transpose_to_bf16<<<(HIDDEN * KV_DIM + 255) / 256, 256, 0, stream>>>(Wv, WvT, HIDDEN, KV_DIM);
  transpose_to_bf16<<<(HIDDEN * HIDDEN + 255) / 256, 256, 0, stream>>>(Wo, WoT, HIDDEN, HIDDEN);

  // 2) QKV projections (WMMA, 64x32 tile per wave)
  {
    int tiles = (SEQ / 64) * (HIDDEN / 32);
    gemm_wmma<<<(tiles + 7) / 8, 256, 0, stream>>>(xb, WqT, Qraw, SEQ, HIDDEN, HIDDEN);
  }
  {
    int tiles = (SEQ / 64) * (KV_DIM / 32);
    gemm_wmma<<<(tiles + 7) / 8, 256, 0, stream>>>(xb, WkT, Kraw, SEQ, KV_DIM, HIDDEN);
    gemm_wmma<<<(tiles + 7) / 8, 256, 0, stream>>>(xb, WvT, Vraw, SEQ, KV_DIM, HIDDEN);
  }

  // 3) RMSNorm + RoPE (+ V transpose-convert), head-major padded layouts
  normrope<<<dim3(SEQ, 12), 32, 0, stream>>>(Qraw, Kraw, Vraw, cosb, sinb, qw, kw,
                                             Qb, Kb, Vt);

  // 4) flash attention with SWA + sink (WMMA + dual TDM staging)
  attn_swa<<<dim3(NH, SEQ / 64), 128, 0, stream>>>(Qb, Kb, Vt, AOb);

  // 5) output projection (WMMA) -> f32 d_out
  {
    int tiles = (SEQ / 64) * (HIDDEN / 32);
    gemm_wmma<<<(tiles + 7) / 8, 256, 0, stream>>>(AOb, WoT, out, SEQ, HIDDEN, HIDDEN);
  }
}
